// HNODE_20933670601183
// MI455X (gfx1250) — compile-verified, matrix operand order checked by reading, not run on hardware
//
#include <hip/hip_runtime.h>
#include <math.h>

// ---------------------------------------------------------------------------
// Problem constants (from reference)
// ---------------------------------------------------------------------------
#define BB   16
#define SS   64
#define PP   4
#define JJ   18
#define DIN  2
#define DOUT 3
#define DD   256
#define HH   512

#define CDIV(a,b) (((a)+(b)-1)/(b))

typedef __attribute__((ext_vector_type(16))) __bf16 v16bf;
typedef __attribute__((ext_vector_type(2)))  __bf16 v2bf;
typedef __attribute__((ext_vector_type(8)))  float  v8f;

// Generalized row map: for logical row r (0..M-1) of a matrix view,
//   physRow = (r / nj) * outer + (r % nj) * inner + base
//   element = X[physRow * ld + k]
struct Map { int nj, outer, inner, base, ld; };

__device__ __forceinline__ __bf16 f2bf(float f) { return (__bf16)f; }

// ---------------------------------------------------------------------------
// Weight repack: f32 (K x N, row-major) -> bf16 WMMA B-fragment layout.
// Block (ntile,ktile) stored at ((ntile*Kt + kt) * 512) halves; within a block
// lane L holds 16 contiguous halves: n = nt*16 + (L&15), kbase = (L>>4)*8,
// halves h<8 -> k = kbase+h ; h>=8 -> k = kbase+16+(h-8). Zero padded OOB.
// ---------------------------------------------------------------------------
__global__ void k_pack(const float* __restrict__ W, __bf16* __restrict__ Pk,
                       int K, int N) {
    int Kt = (K + 31) >> 5, Nt = (N + 15) >> 4;
    long total = (long)Kt * Nt * 512;
    long tid = (long)blockIdx.x * 256 + threadIdx.x;
    if (tid >= total) return;
    int  within = (int)(tid & 511);
    long blk    = tid >> 9;
    int  kt = (int)(blk % Kt);
    int  nt = (int)(blk / Kt);
    int  lane = within >> 4, h = within & 15;
    int  n  = nt * 16 + (lane & 15);
    int  kb = (lane >> 4) << 3;
    int  k  = (kt << 5) + kb + (h < 8 ? h : 8 + h);
    float v = (k < K && n < N) ? W[(long)k * N + n] : 0.0f;
    Pk[tid] = f2bf(v);
}

// ---------------------------------------------------------------------------
// Generic fused linear: Y = epilogue(X @ W + bias)
//   mode 0: identity   mode 1: tanh   mode 2: Y = Base + (X@W+b)*scale
// One workgroup = 8 waves = one 16-row M-tile x eight 16-col N-tiles.
// Per 32-K slab the block cooperatively stages the 16x32 f32 A tile into LDS
// in WMMA A-fragment order (bf16, double-buffered). Software-pipelined:
// slab kt+1's A f32 pair + B fragment are issued before slab kt's barrier so
// their latency hides under the barrier + WMMA. The main loop is branch-free
// (EXEC stays all-ones as WMMA requires); K-tails (K=2, 258) take one guarded
// epilogue slab; out-of-range tiles are clamped and discarded at the store.
// ---------------------------------------------------------------------------
__global__ __launch_bounds__(256)
void k_linear(const float* __restrict__ X, Map mx,
              const __bf16* __restrict__ Wp, const float* __restrict__ bias,
              float* __restrict__ Y, Map my,
              const float* __restrict__ Base, float scale, int mode,
              int M, int K, int N) {
    __shared__ __attribute__((aligned(64))) __bf16 sA[2][512];

    const int Kt  = (K + 31) >> 5;   // total slabs (incl. tail)
    const int KtF = K >> 5;          // unguarded full slabs
    const int Nt  = (N + 15) >> 4;
    const int mt  = blockIdx.x / CDIV(Nt, 8);
    const int wave = threadIdx.x >> 5;
    const int nt  = (blockIdx.x % CDIV(Nt, 8)) * 8 + wave;
    const int ntc = (nt < Nt) ? nt : Nt - 1;   // clamp: run WMMA, drop at store
    const int lane = threadIdx.x & 31;

    // --- cooperative A staging: thread -> (row ml) x (k-pair kp,kp+1) ---
    const int ml = threadIdx.x >> 4;          // 0..15
    const int kp = (threadIdx.x & 15) << 1;   // 0,2,..,30 (coalesced per row)
    int r = mt * 16 + ml; if (r >= M) r = M - 1;
    const long xrow = (long)(r / mx.nj) * mx.outer + (long)(r % mx.nj) * mx.inner + mx.base;
    const float* xr = X + xrow * mx.ld;
    // A-fragment LDS index: lane = ml + (kp&8 ? 16:0), half = (kp&7) + (kp&16 ? 8:0)
    // kp even -> the (kp,kp+1) pair occupies consecutive halves of one lane.
    const int lidx = (ml + ((kp & 8) ? 16 : 0)) * 16 + (kp & 7) + ((kp & 16) ? 8 : 0);

    const __bf16* wblk = Wp + ((long)ntc * Kt) * 512 + lane * 16;

    v8f acc = {};
    float2 fcur = {};
    v16bf  bcur = {};
    if (KtF > 0) {
        fcur = *reinterpret_cast<const float2*>(xr + kp);
        bcur = *reinterpret_cast<const v16bf*>(wblk);
    }
    #pragma unroll 2
    for (int kt = 0; kt < KtF; ++kt) {
        const int buf = kt & 1;
        v2bf pr; pr.x = f2bf(fcur.x); pr.y = f2bf(fcur.y);
        *reinterpret_cast<v2bf*>(&sA[buf][lidx]) = pr;
        // issue next slab's loads before the barrier (latency hidden)
        const int ktn = (kt + 1 < KtF) ? kt + 1 : kt;
        float2 fnx = *reinterpret_cast<const float2*>(xr + (ktn << 5) + kp);
        v16bf  bnx = *reinterpret_cast<const v16bf*>(wblk + (long)ktn * 512);
        __syncthreads();
        v16bf afrag = *reinterpret_cast<const v16bf*>(&sA[buf][lane * 16]);
        acc = __builtin_amdgcn_wmma_f32_16x16x32_bf16(
                  false, afrag, false, bcur, (short)0, acc, false, false);
        fcur = fnx; bcur = bnx;
    }
    if (KtF < Kt) {                           // guarded tail slab
        const int kt = KtF, buf = kt & 1;
        const int k = (kt << 5) + kp;
        float a0 = (k     < K) ? xr[k]     : 0.0f;
        float a1 = (k + 1 < K) ? xr[k + 1] : 0.0f;
        v2bf pr; pr.x = f2bf(a0); pr.y = f2bf(a1);
        *reinterpret_cast<v2bf*>(&sA[buf][lidx]) = pr;
        v16bf bfrag = *reinterpret_cast<const v16bf*>(wblk + (long)kt * 512);
        __syncthreads();
        v16bf afrag = *reinterpret_cast<const v16bf*>(&sA[buf][lane * 16]);
        acc = __builtin_amdgcn_wmma_f32_16x16x32_bf16(
                  false, afrag, false, bfrag, (short)0, acc, false, false);
    }

    // C/D layout: lane -> n = lane&15 ; vgpr i -> m = i + (lane>=16 ? 8 : 0)
    const int n = nt * 16 + (lane & 15);
    if (n >= N) return;
    const float bv = bias ? bias[n] : 0.0f;
    const int mhalf = (lane >> 4) << 3;
    #pragma unroll
    for (int i = 0; i < 8; ++i) {
        int m = mt * 16 + i + mhalf;
        if (m >= M) continue;
        float v = acc[i] + bv;
        if (mode == 1) v = tanhf(v);
        const long yrow = (long)(m / my.nj) * my.outer + (long)(m % my.nj) * my.inner + my.base;
        if (mode == 2) v = Base[yrow * my.ld + n] + v * scale;
        Y[yrow * my.ld + n] = v;
    }
}

// ---------------------------------------------------------------------------
// Elementwise helpers
// ---------------------------------------------------------------------------
__global__ void k_mask(const float* __restrict__ x2d, const float* __restrict__ mask,
                       float* __restrict__ xs, int n) {
    int t = blockIdx.x * 256 + threadIdx.x;
    if (t < n) xs[t] = x2d[t] * mask[t >> 1];
}

__global__ void k_inith(float* __restrict__ h, const float* __restrict__ h0, int n) {
    int t = blockIdx.x * 256 + threadIdx.x;
    if (t < n) h[t] = h0[t & (DD - 1)];
}

__global__ void k_copy(float* __restrict__ d, const float* __restrict__ s, int n) {
    int t = blockIdx.x * 256 + threadIdx.x;
    if (t < n) d[t] = s[t];
}

// Build concat rows: dst[r, 0:Ka] = srcA[joint jaoff + (r%nj)/jadiv],
//                    dst[r, Ka:Ka+Kb] = srcB[joint jboff + (r%nj)]  (or xs slice)
__global__ void k_cat(float* __restrict__ dst, int ldc, int M, int nj,
                      const float* __restrict__ srcA, int jaoff, int jadiv, int Ka,
                      const float* __restrict__ srcB, int jboff, int Kb,
                      const float* __restrict__ xs, int t, int useXs) {
    long tid = (long)blockIdx.x * 256 + threadIdx.x;
    if (tid >= (long)M * ldc) return;
    int r = (int)(tid / ldc), c = (int)(tid % ldc);
    int bp = r / nj, jr = r % nj;
    float v = 0.0f;
    if (c < Ka) {
        int j = jaoff + jr / jadiv;
        v = srcA[((long)(bp * JJ + j)) * DD + c];
    } else if (c < Ka + Kb) {
        int cb = c - Ka, j = jboff + jr;
        if (useXs) {
            int b = bp >> 2, p = bp & 3;
            v = xs[((((long)b * SS + t) * PP + p) * JJ + j) * DIN + cb];
        } else {
            v = srcB[((long)(bp * JJ + j)) * DD + cb];
        }
    }
    dst[tid] = v;
}

// GRU gate combine: gates gi/gh are dense (M x 768); hidden h1 and output hn
// are (nj,joff) group slices of full (B*P*J, 256) state buffers.
__global__ void k_gru(const float* __restrict__ gi, const float* __restrict__ gh,
                      const float* __restrict__ h1, float* __restrict__ hn,
                      int nj, int joff, int M) {
    int tid = blockIdx.x * 256 + threadIdx.x;
    if (tid >= M * DD) return;
    int r = tid >> 8, d = tid & 255;
    long row = (long)(r / nj) * JJ + joff + (r % nj);
    const float* gir = gi + (long)r * (3 * DD);
    const float* ghr = gh + (long)r * (3 * DD);
    float ir = gir[d], iz = gir[DD + d], in = gir[2 * DD + d];
    float hr = ghr[d], hz = ghr[DD + d], hv = ghr[2 * DD + d];
    float h  = h1[row * DD + d];
    float rg = 1.0f / (1.0f + expf(-(ir + hr)));
    float zg = 1.0f / (1.0f + expf(-(iz + hz)));
    float ng = tanhf(in + rg * hv);
    hn[row * DD + d] = (1.0f - zg) * ng + zg * h;
}

// Output projection: out[b,t,p,j,:] = h[(b*P+p)*J+j] @ Wo(256x3) + bo
__global__ void k_out(const float* __restrict__ h, const float* __restrict__ Wo,
                      const float* __restrict__ bo, float* __restrict__ out, int t) {
    int tid = blockIdx.x * 256 + threadIdx.x;
    if (tid >= BB * PP * JJ * DOUT) return;
    int r = tid / DOUT, n = tid % DOUT;
    const float* hp = h + (long)r * DD;
    float s = bo[n];
    #pragma unroll 4
    for (int k = 0; k < DD; ++k) s += hp[k] * Wo[k * DOUT + n];
    int b = r / (PP * JJ), rem = r % (PP * JJ), p = rem / JJ, j = rem % JJ;
    out[((((long)b * SS + t) * PP + p) * JJ + j) * DOUT + n] = s;
}

// ---------------------------------------------------------------------------
// Host orchestration
// ---------------------------------------------------------------------------
static inline Map mapDense(int ld)          { Map m{1, 1, 0, 0, ld};       return m; }
static inline Map mapH(int nj, int joff)    { Map m{nj, JJ, 1, joff, DD};  return m; }
static inline Map mapXs(int t)              { Map m{PP, SS*PP*JJ, JJ, t*PP*JJ, DIN}; return m; }

static inline void launch_linear(hipStream_t s, const float* X, Map mx,
                                 const __bf16* Wp, const float* b,
                                 float* Y, Map my, const float* Base, float scale,
                                 int mode, int M, int K, int N) {
    int Mt = CDIV(M, 16), Nt = CDIV(N, 16);
    int blocks = Mt * CDIV(Nt, 8);
    k_linear<<<blocks, 256, 0, s>>>(X, mx, Wp, b, Y, my, Base, scale, mode, M, K, N);
}

struct PD { int idx; int K; int N; };
// d_in layout assumption: jax pytree flatten (dict keys sorted):
// 0 ts, 1 mask, 2 x2d, 3 g1, 4 g2, 5 g3,
// GRU_k blocks each as {Wh,Wi,bh,bi}: GRU_1 6..13, GRU_2 14..21, GRU_3 22..29,
// ODE_k as [W0,b0,W1,b1,W2,b2]: ODE_1 30..35, ODE_2 36..41, ODE_3 42..47,
// 48 h0, 49 out.W, 50 out.b
static const PD g_pd[21] = {
    {30,256,512},{32,512,512},{34,512,256},        // ODE_1
    {36,512,512},{38,512,512},{40,512,256},        // ODE_2
    {42,512,512},{44,512,512},{46,512,256},        // ODE_3
    {7,  2,768},{6,256,768},{11,256,768},{10,256,768},   // GRU_1 Wi0,Wh0,Wi1,Wh1
    {15,258,768},{14,256,768},{19,256,768},{18,256,768}, // GRU_2
    {23,258,768},{22,256,768},{27,256,768},{26,256,768}, // GRU_3
};

extern "C" void kernel_launch(void* const* d_in, const int* in_sizes, int n_in,
                              void* d_out, int out_size, void* d_ws, size_t ws_size,
                              hipStream_t stream) {
    (void)in_sizes; (void)n_in; (void)out_size; (void)ws_size;
    auto Pf = [&](int i) { return (const float*)d_in[i]; };
    float* out = (float*)d_out;

    // --- workspace carve-up (floats) ---
    const int NX   = BB * SS * PP * JJ * DIN;   // 147456
    const int NH_  = BB * PP * JJ * DD;         // 294912
    const int NCAT = 768 * 512;
    const int NHID = 768 * 512;
    const int NG   = 768 * 768;
    float* ws  = (float*)d_ws;  size_t off = 0;
    float* xs   = ws + off; off += NX;
    float* hA   = ws + off; off += NH_;
    float* hB   = ws + off; off += NH_;
    float* h1b  = ws + off; off += NH_;
    float* cat  = ws + off; off += NCAT;
    float* hid1 = ws + off; off += NHID;
    float* hid2 = ws + off; off += NHID;
    float* gi   = ws + off; off += NG;
    float* gh   = ws + off; off += NG;
    __bf16* packBase = (__bf16*)(ws + off);

    size_t poff[21]; size_t pacc = 0;
    for (int i = 0; i < 21; ++i) {
        poff[i] = pacc;
        pacc += (size_t)CDIV(g_pd[i].K, 32) * CDIV(g_pd[i].N, 16) * 512;
    }
    auto PW = [&](int i) { return (const __bf16*)(packBase + poff[i]); };

    // --- one-time prep (re-done every call: deterministic) ---
    k_mask<<<CDIV(NX,256),256,0,stream>>>(Pf(2), Pf(1), xs, NX);
    k_inith<<<CDIV(NH_,256),256,0,stream>>>(hA, Pf(48), NH_);
    for (int i = 0; i < 21; ++i) {
        long tot = (long)CDIV(g_pd[i].K,32) * CDIV(g_pd[i].N,16) * 512;
        k_pack<<<(int)CDIV(tot,256),256,0,stream>>>(Pf(g_pd[i].idx),
                                                    packBase + poff[i],
                                                    g_pd[i].K, g_pd[i].N);
    }

    float* hcur = hA;  float* hnext = hB;
    for (int t = 0; t < SS; ++t) {
        // h1 <- h (ODE state)
        k_copy<<<CDIV(NH_,256),256,0,stream>>>(h1b, hcur, NH_);

        // ODE_1 (group g1: joint 0, M=64): h1_1 += MLP(h1_1) * 2k*dt
        launch_linear(stream, h1b, mapH(1,0), PW(0), Pf(31), hid1, mapDense(512), nullptr,0,1, 64,256,512);
        launch_linear(stream, hid1, mapDense(512), PW(1), Pf(33), hid2, mapDense(512), nullptr,0,1, 64,512,512);
        launch_linear(stream, hid2, mapDense(512), PW(2), Pf(35), h1b, mapH(1,0), h1b, 0.08f,2, 64,512,256);

        for (int i = 0; i < 2; ++i) {
            // ODE_2 (g2: joints 1..5, M=320), input = [bcast h1_1 | h1_2]
            k_cat<<<CDIV(320*512,256),256,0,stream>>>(cat,512,320,5, h1b,0,8,256, h1b,1,256, nullptr,0,0);
            launch_linear(stream, cat, mapDense(512), PW(3), Pf(37), hid1, mapDense(512), nullptr,0,1, 320,512,512);
            launch_linear(stream, hid1, mapDense(512), PW(4), Pf(39), hid2, mapDense(512), nullptr,0,1, 320,512,512);
            launch_linear(stream, hid2, mapDense(512), PW(5), Pf(41), h1b, mapH(5,1), h1b, 0.04f,2, 320,512,256);
            for (int j = 0; j < 2; ++j) {
                // ODE_3 (g3: joints 6..17, M=768), input = [bcast h1_2[1+gi] | h1_3]
                k_cat<<<CDIV(768*512,256),256,0,stream>>>(cat,512,768,12, h1b,2,3,256, h1b,6,256, nullptr,0,0);
                launch_linear(stream, cat, mapDense(512), PW(6), Pf(43), hid1, mapDense(512), nullptr,0,1, 768,512,512);
                launch_linear(stream, hid1, mapDense(512), PW(7), Pf(45), hid2, mapDense(512), nullptr,0,1, 768,512,512);
                launch_linear(stream, hid2, mapDense(512), PW(8), Pf(47), h1b, mapH(12,6), h1b, 0.02f,2, 768,512,256);
            }
        }

        // GRU_1 (M=64, hidden h1 joint 0)
        launch_linear(stream, xs, mapXs(t), PW(9),  Pf(9),  gi, mapDense(768), nullptr,0,0, 64,2,768);
        launch_linear(stream, h1b, mapH(1,0), PW(10), Pf(8), gh, mapDense(768), nullptr,0,0, 64,256,768);
        k_gru<<<CDIV(64*DD,256),256,0,stream>>>(gi, gh, h1b, hnext, 1,0, 64);
        launch_linear(stream, hnext, mapH(1,0), PW(11), Pf(13), gi, mapDense(768), nullptr,0,0, 64,256,768);
        launch_linear(stream, h1b,   mapH(1,0), PW(12), Pf(12), gh, mapDense(768), nullptr,0,0, 64,256,768);
        k_gru<<<CDIV(64*DD,256),256,0,stream>>>(gi, gh, h1b, hnext, 1,0, 64);

        // GRU_2 (M=320), input0 = [bcast h2_1 | x_2] (K=258, ld padded to 264)
        k_cat<<<CDIV(320*264,256),256,0,stream>>>(cat,264,320,5, hnext,0,8,256, nullptr,1,2, xs,t,1);
        launch_linear(stream, cat, mapDense(264), PW(13), Pf(17), gi, mapDense(768), nullptr,0,0, 320,258,768);
        launch_linear(stream, h1b, mapH(5,1),     PW(14), Pf(16), gh, mapDense(768), nullptr,0,0, 320,256,768);
        k_gru<<<CDIV(320*DD,256),256,0,stream>>>(gi, gh, h1b, hnext, 5,1, 320);
        launch_linear(stream, hnext, mapH(5,1), PW(15), Pf(21), gi, mapDense(768), nullptr,0,0, 320,256,768);
        launch_linear(stream, h1b,   mapH(5,1), PW(16), Pf(20), gh, mapDense(768), nullptr,0,0, 320,256,768);
        k_gru<<<CDIV(320*DD,256),256,0,stream>>>(gi, gh, h1b, hnext, 5,1, 320);

        // GRU_3 (M=768), input0 = [bcast h2_2[1+gi] | x_3]
        k_cat<<<CDIV(768*264,256),256,0,stream>>>(cat,264,768,12, hnext,2,3,256, nullptr,6,2, xs,t,1);
        launch_linear(stream, cat, mapDense(264), PW(17), Pf(25), gi, mapDense(768), nullptr,0,0, 768,258,768);
        launch_linear(stream, h1b, mapH(12,6),    PW(18), Pf(24), gh, mapDense(768), nullptr,0,0, 768,256,768);
        k_gru<<<CDIV(768*DD,256),256,0,stream>>>(gi, gh, h1b, hnext, 12,6, 768);
        launch_linear(stream, hnext, mapH(12,6), PW(19), Pf(29), gi, mapDense(768), nullptr,0,0, 768,256,768);
        launch_linear(stream, h1b,   mapH(12,6), PW(20), Pf(28), gh, mapDense(768), nullptr,0,0, 768,256,768);
        k_gru<<<CDIV(768*DD,256),256,0,stream>>>(gi, gh, h1b, hnext, 12,6, 768);

        // Output projection for this step
        k_out<<<CDIV(BB*PP*JJ*DOUT,256),256,0,stream>>>(hnext, Pf(49), Pf(50), out, t);

        float* tmp = hcur; hcur = hnext; hnext = tmp;
    }
}
